// RichRepresentativePointConstructor_32882269618901
// MI455X (gfx1250) — matrix operand order repaired and checked by reading, not compiled, not run on hardware
//
#include <hip/hip_runtime.h>
#include <hip/hip_bf16.h>
#include <stdint.h>

// ---------------- problem constants ----------------
#define NB 16
#define NT 2048
#define NK 128
#define NP (NB * NK)   // 2048 points
#define DV 768
#define DC 256
#define HH 128         // LSTM hidden per direction
#define GW 512         // 4*H gates per direction
#define WW 11          // window length (2*CW+1)
#define CWc 5

typedef unsigned short u16;
typedef __attribute__((ext_vector_type(16))) __bf16 v16bf;
typedef __attribute__((ext_vector_type(8)))  float  v8f;

union Frag { v16bf v; uint4 q[2]; };

static __device__ __forceinline__ u16 f32_to_bf16(float f) {
  union { float f; unsigned int u; } x; x.f = f;
  unsigned int u = x.u;
  u += 0x7FFFu + ((u >> 16) & 1u);          // round-to-nearest-even
  return (u16)(u >> 16);
}
static __device__ __forceinline__ float bf16_to_f32(u16 h) {
  union { float f; unsigned int u; } x; x.u = ((unsigned int)h) << 16;
  return x.f;
}
static __device__ __forceinline__ float sigmoidf_(float x) {
  return 1.f / (1.f + __expf(-x));
}

// ---------------- weight prep ----------------
// f32 -> bf16 row copy with optional zero padding of trailing columns.
__global__ void cvt_pad_kernel(const float* __restrict__ src, u16* __restrict__ dst,
                               int rows, int scols, int dcols) {
  size_t n = (size_t)rows * dcols;
  for (size_t i = (size_t)blockIdx.x * blockDim.x + threadIdx.x; i < n;
       i += (size_t)gridDim.x * blockDim.x) {
    int r = (int)(i / dcols), c = (int)(i % dcols);
    dst[i] = (c < scols) ? f32_to_bf16(src[(size_t)r * scols + c]) : (u16)0;
  }
}

__global__ void bias_cat_kernel(const float* bihf, const float* bhhf,
                                const float* bihb, const float* bhhb, float* out) {
  int i = blockIdx.x * blockDim.x + threadIdx.x;
  if (i < GW)      out[i]       = bihf[i] + bhhf[i];
  else if (i < 2*GW) { int j = i - GW; out[i] = bihb[j] + bhhb[j]; }
}

__global__ void zero_u16_kernel(u16* p, size_t n) {
  for (size_t i = (size_t)blockIdx.x * blockDim.x + threadIdx.x; i < n;
       i += (size_t)gridDim.x * blockDim.x) p[i] = 0;
}

// ---------------- window gather (masked, bf16, 16-row padded) ----------------
__global__ __launch_bounds__(256) void gather_kernel(const float* __restrict__ video,
                                                     const int* __restrict__ idx,
                                                     u16* __restrict__ Xg) {
  const int pt = blockIdx.x;            // 0..2047
  const int b  = pt >> 7;
  const int id = idx[pt];
  const int start = id - CWc > 0 ? id - CWc : 0;
  const int wl    = (id + CWc + 1 < NT ? id + CWc + 1 : NT) - start;
  for (int e = threadIdx.x; e < 16 * DV; e += 256) {
    const int w = e / DV, d = e - w * DV;
    float v = 0.f;
    if (w < WW && w < wl) {
      int pos = start + w; if (pos > NT - 1) pos = NT - 1;
      v = video[((size_t)b * NT + pos) * DV + d];
    }
    Xg[(size_t)pt * 16 * DV + e] = f32_to_bf16(v);
  }
}

// ---------------- generic bf16 WMMA GEMM: C = act(A @ B^T + bias) ----------------
// A: (rows, lda) bf16 row-major.  B: (cols, ldb) bf16 row-major (weights).
// grid.x = M/16, grid.y = N/64; 4 waves/block, one 16x16 tile per wave.
__global__ __launch_bounds__(128) void wmma_gemm_kernel(
    const u16* __restrict__ A, int lda,
    const u16* __restrict__ B, int ldb,
    const float* __restrict__ bias,
    float* __restrict__ Cf, u16* __restrict__ Cb, int ldc,
    int K, int act) {
  const int tid  = threadIdx.x;
  const int wave = tid >> 5;
  const int lane = tid & 31;
  const int lr   = lane & 15;
  const int hi   = lane >> 4;
  const int m0 = blockIdx.x * 16;
  const int n0 = (blockIdx.y * 4 + wave) * 16;
  const u16* arow = A + (size_t)(m0 + lr) * lda + hi * 8;    // A frag: 2x16B runs
  const u16* brow = B + (size_t)(n0 + lr) * ldb + hi * 8;    // B frag: 2x16B runs
  v8f acc = {0.f,0.f,0.f,0.f,0.f,0.f,0.f,0.f};
  for (int k0 = 0; k0 < K; k0 += 32) {
    Frag a, b;
    a.q[0] = *reinterpret_cast<const uint4*>(arow + k0);
    a.q[1] = *reinterpret_cast<const uint4*>(arow + k0 + 16);
    b.q[0] = *reinterpret_cast<const uint4*>(brow + k0);
    b.q[1] = *reinterpret_cast<const uint4*>(brow + k0 + 16);
    acc = __builtin_amdgcn_wmma_f32_16x16x32_bf16(false, a.v, false, b.v,
                                                  (short)0, acc, false, false);
  }
  const float bv = bias ? bias[n0 + lr] : 0.f;
  for (int r = 0; r < 8; ++r) {
    float v = acc[r] + bv;
    if (act == 1) v = v > 0.f ? v : 0.f;
    const size_t off = (size_t)(m0 + r + hi * 8) * ldc + (n0 + lr);
    if (Cf) Cf[off] = v;
    else    Cb[off] = f32_to_bf16(v);
  }
}

// ---------------- batched LSTM scan: 16 points per workgroup ----------------
// Pre: (NP,16,1024) f32; Whh: (512,128) bf16; per step: G = Pre[t] + H @ Whh^T (WMMA).
__global__ __launch_bounds__(256) void lstm_scan_kernel(
    const float* __restrict__ Pre, const u16* __restrict__ Whh,
    int gofs, int rev,
    u16* __restrict__ Hseq, int dircol,      // if non-null: store all t (N,16,256)
    float* __restrict__ ctx) {               // if non-null: store last h (N,256) cols 0..127
  __shared__ float Gs[16 * GW];              // 32 KB
  __shared__ u16   Hc[16 * HH];              // 4 KB
  const int tid  = threadIdx.x;
  const int wave = tid >> 5;
  const int lane = tid & 31;
  const int lr   = lane & 15;
  const int hi   = lane >> 4;
  const int ptbase = blockIdx.x * 16;
  for (int i = tid; i < 16 * HH; i += 256) Hc[i] = 0;
  float cst[8];
  for (int u = 0; u < 8; ++u) cst[u] = 0.f;
  for (int s = 0; s < WW; ++s) {
    const int t = rev ? (WW - 1 - s) : s;
    __syncthreads();                          // Hc ready
    for (int jt = 0; jt < 4; ++jt) {
      const int n0 = wave * 64 + jt * 16;
      v8f acc;
      for (int r = 0; r < 8; ++r)
        acc[r] = Pre[(size_t)(ptbase + r + hi * 8) * 16384 + (size_t)t * 1024 + gofs + n0 + lr];
      const u16* brow = Whh + (size_t)(n0 + lr) * HH + hi * 8;
      const u16* arow = &Hc[lr * HH + hi * 8];
      for (int k0 = 0; k0 < HH; k0 += 32) {
        Frag a, b;
        a.q[0] = *reinterpret_cast<const uint4*>(arow + k0);
        a.q[1] = *reinterpret_cast<const uint4*>(arow + k0 + 16);
        b.q[0] = *reinterpret_cast<const uint4*>(brow + k0);
        b.q[1] = *reinterpret_cast<const uint4*>(brow + k0 + 16);
        acc = __builtin_amdgcn_wmma_f32_16x16x32_bf16(false, a.v, false, b.v,
                                                      (short)0, acc, false, false);
      }
      for (int r = 0; r < 8; ++r)
        Gs[(r + hi * 8) * GW + n0 + lr] = acc[r];
    }
    __syncthreads();                          // Gs ready, Hc reads done
    for (int u = 0; u < 8; ++u) {
      const int lin = tid * 8 + u;
      const int p = lin >> 7, j = lin & 127;
      const float gi = Gs[p * GW + j];
      const float gf = Gs[p * GW + 128 + j];
      const float gg = Gs[p * GW + 256 + j];
      const float go = Gs[p * GW + 384 + j];
      cst[u] = sigmoidf_(gf) * cst[u] + sigmoidf_(gi) * tanhf(gg);
      const float h = sigmoidf_(go) * tanhf(cst[u]);
      const u16 hb = f32_to_bf16(h);
      Hc[p * HH + j] = hb;
      if (Hseq) Hseq[(size_t)(ptbase + p) * 4096 + (size_t)t * 256 + dircol * HH + j] = hb;
      if (ctx && s == WW - 1) ctx[(size_t)(ptbase + p) * 256 + j] = h;
    }
  }
}

// layer-2 backward: output at [:, -1] is scan step 0 (h0=c0=0) -> single cell on t=10.
__global__ void l2b_cell_kernel(const float* __restrict__ Pre, float* __restrict__ ctx) {
  const int lin = blockIdx.x * blockDim.x + threadIdx.x;
  if (lin >= NP * HH) return;
  const int pt = lin >> 7, j = lin & 127;
  const float* g = Pre + (size_t)pt * 16384 + 10 * 1024 + GW;
  const float c = sigmoidf_(g[j]) * tanhf(g[256 + j]);
  const float h = sigmoidf_(g[384 + j]) * tanhf(c);
  ctx[(size_t)pt * 256 + 128 + j] = h;
}

// ---------------- heads ----------------
__global__ void comb_build_kernel(const float* __restrict__ video, const int* __restrict__ idx,
                                  const float* __restrict__ ctx, u16* __restrict__ comb) {
  const int pt = blockIdx.x, b = pt >> 7;
  const int id = idx[pt];
  for (int e = threadIdx.x; e < 1024; e += 128) {
    float v = (e < DV) ? video[((size_t)b * NT + id) * DV + e]
                       : ctx[(size_t)pt * 256 + e - DV];
    comb[(size_t)pt * 1024 + e] = f32_to_bf16(v);
  }
}

__global__ __launch_bounds__(256) void ln_relu_kernel(const float* __restrict__ X,
                                                      const float* __restrict__ g,
                                                      const float* __restrict__ bn,
                                                      u16* __restrict__ Y, int D) {
  __shared__ float s1[256], s2[256];
  const int row = blockIdx.x, tid = threadIdx.x;
  const float* x = X + (size_t)row * D;
  float a = 0.f, b = 0.f;
  for (int i = tid; i < D; i += 256) { float v = x[i]; a += v; b += v * v; }
  s1[tid] = a; s2[tid] = b; __syncthreads();
  for (int o = 128; o > 0; o >>= 1) {
    if (tid < o) { s1[tid] += s1[tid + o]; s2[tid] += s2[tid + o]; }
    __syncthreads();
  }
  const float mean = s1[0] / D;
  const float var  = s2[0] / D - mean * mean;
  const float rstd = rsqrtf(var + 1e-5f);
  for (int i = tid; i < D; i += 256) {
    float v = (x[i] - mean) * rstd * g[i] + bn[i];
    v = v > 0.f ? v : 0.f;
    Y[(size_t)row * D + i] = f32_to_bf16(v);
  }
}

// dot(384) + sigmoid (rep.W2 / imp.W2 heads)
__global__ void dot384_sig_kernel(const u16* __restrict__ H, const float* __restrict__ W,
                                  const float* __restrict__ b2, float* __restrict__ out) {
  const int pt = blockIdx.x * blockDim.x + threadIdx.x;
  if (pt >= NP) return;
  float s = b2[0];
  for (int j = 0; j < 384; ++j) s += bf16_to_f32(H[(size_t)pt * 384 + j]) * W[j];
  out[pt] = sigmoidf_(s);
}

__global__ void softmax_kernel(const float* __restrict__ wlog, float* __restrict__ wts) {
  __shared__ float sh[128];
  const int b = blockIdx.x, k = threadIdx.x;
  const float v = wlog[b * NK + k];                 // sigmoid already applied upstream
  sh[k] = v; __syncthreads();
  for (int o = 64; o > 0; o >>= 1) { if (k < o) sh[k] = fmaxf(sh[k], sh[k + o]); __syncthreads(); }
  const float mx = sh[0]; __syncthreads();
  const float e = __expf(v - mx);
  sh[k] = e; __syncthreads();
  for (int o = 64; o > 0; o >>= 1) { if (k < o) sh[k] += sh[k + o]; __syncthreads(); }
  wts[b * NK + k] = e / sh[0];
}

__global__ void cov2_kernel(const u16* __restrict__ H, const float* __restrict__ W2,
                            const float* __restrict__ b2, const int* __restrict__ idx,
                            float* __restrict__ span) {
  const int pt = blockIdx.x * blockDim.x + threadIdx.x;
  if (pt >= NP) return;
  float o[2];
  for (int c = 0; c < 2; ++c) {
    float s = b2[c];
    for (int j = 0; j < 64; ++j) s += bf16_to_f32(H[(size_t)pt * 64 + j]) * W2[c * 64 + j];
    o[c] = tanhf(s);
  }
  const float center = (float)idx[pt];
  const float c0 = fminf(fmaxf(center + o[0] * (float)(NT / 4), 0.f), (float)(NT - 1));
  const float c1 = fminf(fmaxf(center + o[1] * (float)(NT / 4), 0.f), (float)(NT - 1));
  span[pt] = (c1 - c0) / (float)NT;
}

__global__ void gpe_kernel(const int* __restrict__ idx, const float* __restrict__ W1,
                           const float* __restrict__ b1, const float* __restrict__ W2,
                           const float* __restrict__ b2, float* __restrict__ gtmp) {
  const int pt = blockIdx.x * blockDim.x + threadIdx.x;
  if (pt >= NP) return;
  const float x = (float)idx[pt] / (float)(NT - 1);   // linspace(0,1,T)[idx]
  float h[32];
  for (int i = 0; i < 32; ++i) { float v = W1[i] * x + b1[i]; h[i] = v > 0.f ? v : 0.f; }
  for (int o = 0; o < 64; ++o) {
    float s = b2[o];
    for (int i = 0; i < 32; ++i) s += W2[o * 32 + i] * h[i];
    gtmp[(size_t)pt * 64 + o] = s;
  }
}

// fin = [comb(1024) | g_tmp(64) | rel_emb row (64) | wts | imp | span | 0-pad -> 1184]
__global__ void fin_build_kernel(const u16* __restrict__ comb, const float* __restrict__ gtmp,
                                 const float* __restrict__ rel_emb, const float* __restrict__ wts,
                                 const float* __restrict__ impv, const float* __restrict__ span,
                                 u16* __restrict__ fin) {
  const int pt = blockIdx.x, k = pt & 127;
  const int relrow = (k == 0) ? CWc : (CWc + 1);      // min(CW+1, 2*CW) = 6
  for (int e = threadIdx.x; e < 1184; e += 128) {
    u16 v;
    if (e < 1024)       v = comb[(size_t)pt * 1024 + e];
    else if (e < 1088)  v = f32_to_bf16(gtmp[(size_t)pt * 64 + (e - 1024)]);
    else if (e < 1152)  v = f32_to_bf16(rel_emb[relrow * 64 + (e - 1088)]);
    else if (e == 1152) v = f32_to_bf16(wts[pt]);
    else if (e == 1153) v = f32_to_bf16(impv[pt]);
    else if (e == 1154) v = f32_to_bf16(span[pt]);
    else                v = 0;
    fin[(size_t)pt * 1184 + e] = v;
  }
}

// ---------------- host launch ----------------
extern "C" void kernel_launch(void* const* d_in, const int* in_sizes, int n_in,
                              void* d_out, int out_size, void* d_ws, size_t ws_size,
                              hipStream_t stream) {
  (void)in_sizes; (void)n_in; (void)out_size; (void)ws_size;
  // Input leaves in setup_inputs() insertion order (depth-first through nested dicts).
  const float* video = (const float*)d_in[0];
  const float* l1fWih = (const float*)d_in[1];  const float* l1fWhh = (const float*)d_in[2];
  const float* l1fbih = (const float*)d_in[3];  const float* l1fbhh = (const float*)d_in[4];
  const float* l1bWih = (const float*)d_in[5];  const float* l1bWhh = (const float*)d_in[6];
  const float* l1bbih = (const float*)d_in[7];  const float* l1bbhh = (const float*)d_in[8];
  const float* l2fWih = (const float*)d_in[9];  const float* l2fWhh = (const float*)d_in[10];
  const float* l2fbih = (const float*)d_in[11]; const float* l2fbhh = (const float*)d_in[12];
  const float* l2bWih = (const float*)d_in[13]; const float* l2bWhh = (const float*)d_in[14];
  const float* l2bbih = (const float*)d_in[15]; const float* l2bbhh = (const float*)d_in[16];
  const float* gpeW1 = (const float*)d_in[17];  const float* gpeb1 = (const float*)d_in[18];
  const float* gpeW2 = (const float*)d_in[19];  const float* gpeb2 = (const float*)d_in[20];
  const float* rel_emb = (const float*)d_in[21];
  const float* repW1 = (const float*)d_in[22];  const float* repb1 = (const float*)d_in[23];
  const float* repW2 = (const float*)d_in[24];  const float* repb2 = (const float*)d_in[25];
  const float* repg  = (const float*)d_in[26];  const float* repbn = (const float*)d_in[27];
  const float* covW1 = (const float*)d_in[28];  const float* covb1 = (const float*)d_in[29];
  const float* covW2 = (const float*)d_in[30];  const float* covb2 = (const float*)d_in[31];
  const float* impW1 = (const float*)d_in[32];  const float* impb1 = (const float*)d_in[33];
  const float* impW2 = (const float*)d_in[34];  const float* impb2 = (const float*)d_in[35];
  const float* fuseW1 = (const float*)d_in[36]; const float* fuseb1 = (const float*)d_in[37];
  const float* fuseW2 = (const float*)d_in[38]; const float* fuseb2 = (const float*)d_in[39];
  const float* fuseg  = (const float*)d_in[40]; const float* fusebn = (const float*)d_in[41];
  const int*   idx    = (const int*)d_in[42];
  float* out = (float*)d_out;

  // workspace carve
  char* wp = (char*)d_ws;
  auto carve = [&](size_t bytes) -> void* {
    void* r = (void*)wp; wp += (bytes + 255) & ~(size_t)255; return r;
  };
  u16* wWih1  = (u16*)carve((size_t)1024 * DV * 2);       // [Wih_f;Wih_b] layer1
  u16* wWih2  = (u16*)carve((size_t)1024 * DC * 2);       // [Wih_f;Wih_b] layer2
  u16* wWhh1f = (u16*)carve((size_t)GW * HH * 2);
  u16* wWhh1b = (u16*)carve((size_t)GW * HH * 2);
  u16* wWhh2f = (u16*)carve((size_t)GW * HH * 2);
  u16* wWhh2b = (u16*)carve((size_t)GW * HH * 2);
  u16* wRep1  = (u16*)carve((size_t)384 * 1024 * 2);
  u16* wCov1  = (u16*)carve((size_t)64 * 1024 * 2);
  u16* wImp1  = (u16*)carve((size_t)384 * DV * 2);
  u16* wFuse1 = (u16*)carve((size_t)DV * 1184 * 2);
  u16* wFuse2 = (u16*)carve((size_t)DV * DV * 2);
  float* bsum1 = (float*)carve(1024 * 4);
  float* bsum2 = (float*)carve(1024 * 4);
  u16*   Xg    = (u16*)carve((size_t)NP * 16 * DV * 2);
  float* Pre   = (float*)carve((size_t)NP * 16 * 1024 * 4);   // reused layer1/layer2
  u16*   Hseq  = (u16*)carve((size_t)NP * 16 * 256 * 2);
  float* ctx   = (float*)carve((size_t)NP * 256 * 4);
  u16*   comb  = (u16*)carve((size_t)NP * 1024 * 2);
  float* reph1f = (float*)carve((size_t)NP * 384 * 4);
  u16*   reph1b = (u16*)carve((size_t)NP * 384 * 2);
  u16*   covh1  = (u16*)carve((size_t)NP * 64 * 2);
  u16*   imph1  = (u16*)carve((size_t)NP * 384 * 2);
  float* wlog   = (float*)carve(NP * 4);
  float* wts    = (float*)carve(NP * 4);
  float* impv   = (float*)carve(NP * 4);
  float* span   = (float*)carve(NP * 4);
  float* gtmp   = (float*)carve((size_t)NP * 64 * 4);
  u16*   fin    = (u16*)carve((size_t)NP * 1184 * 2);
  float* fuse1f = (float*)carve((size_t)NP * DV * 4);
  u16*   fuse1b = (u16*)carve((size_t)NP * DV * 2);

  auto cvt = [&](const float* s, u16* d, int rows, int sc, int dc) {
    int n = rows * dc;
    cvt_pad_kernel<<<(n + 255) / 256, 256, 0, stream>>>(s, d, rows, sc, dc);
  };
  // weight prep
  cvt(l1fWih, wWih1, GW, DV, DV);
  cvt(l1bWih, wWih1 + (size_t)GW * DV, GW, DV, DV);
  cvt(l2fWih, wWih2, GW, DC, DC);
  cvt(l2bWih, wWih2 + (size_t)GW * DC, GW, DC, DC);
  cvt(l1fWhh, wWhh1f, GW, HH, HH);
  cvt(l1bWhh, wWhh1b, GW, HH, HH);
  cvt(l2fWhh, wWhh2f, GW, HH, HH);
  cvt(l2bWhh, wWhh2b, GW, HH, HH);
  cvt(repW1, wRep1, 384, 1024, 1024);
  cvt(covW1, wCov1, 64, 1024, 1024);
  cvt(impW1, wImp1, 384, DV, DV);
  cvt(fuseW1, wFuse1, DV, 1155, 1184);     // K padded 1155 -> 1184 (37*32)
  cvt(fuseW2, wFuse2, DV, DV, DV);
  bias_cat_kernel<<<4, 256, 0, stream>>>(l1fbih, l1fbhh, l1bbih, l1bbhh, bsum1);
  bias_cat_kernel<<<4, 256, 0, stream>>>(l2fbih, l2fbhh, l2bbih, l2bbhh, bsum2);
  zero_u16_kernel<<<4096, 256, 0, stream>>>(Hseq, (size_t)NP * 16 * 256);

  auto gemm = [&](const u16* A, int lda, const u16* B, int ldb, const float* bias,
                  float* Cf, u16* Cb, int ldc, int Mtiles, int N, int K, int act) {
    dim3 grid(Mtiles, N / 64);
    wmma_gemm_kernel<<<grid, 128, 0, stream>>>(A, lda, B, ldb, bias, Cf, Cb, ldc, K, act);
  };

  // ---- BiLSTM ----
  gather_kernel<<<NP, 256, 0, stream>>>(video, idx, Xg);
  // layer-1 pre-activations: (NP*16, 768) @ (1024, 768)^T
  gemm(Xg, DV, wWih1, DV, bsum1, Pre, nullptr, 1024, NP, 1024, DV, 0);
  lstm_scan_kernel<<<NP / 16, 256, 0, stream>>>(Pre, wWhh1f, 0,  0, Hseq, 0, nullptr);
  lstm_scan_kernel<<<NP / 16, 256, 0, stream>>>(Pre, wWhh1b, GW, 1, Hseq, 1, nullptr);
  // layer-2 pre-activations: (NP*16, 256) @ (1024, 256)^T  (Pre reused)
  gemm(Hseq, DC, wWih2, DC, bsum2, Pre, nullptr, 1024, NP, 1024, DC, 0);
  lstm_scan_kernel<<<NP / 16, 256, 0, stream>>>(Pre, wWhh2f, 0, 0, nullptr, 0, ctx);
  l2b_cell_kernel<<<(NP * HH) / 256, 256, 0, stream>>>(Pre, ctx);

  // ---- heads ----
  comb_build_kernel<<<NP, 128, 0, stream>>>(video, idx, ctx, comb);
  // rep: LN(relu after LN) -> scalar sigmoid -> softmax over K
  gemm(comb, 1024, wRep1, 1024, repb1, reph1f, nullptr, 384, NP / 16, 384, 1024, 0);
  ln_relu_kernel<<<NP, 256, 0, stream>>>(reph1f, repg, repbn, reph1b, 384);
  dot384_sig_kernel<<<NP / 256, 256, 0, stream>>>(reph1b, repW2, repb2, wlog);
  softmax_kernel<<<NB, NK, 0, stream>>>(wlog, wts);
  // cov: relu MLP -> tanh offsets -> span
  gemm(comb, 1024, wCov1, 1024, covb1, nullptr, covh1, 64, NP / 16, 64, 1024, 1);
  cov2_kernel<<<NP / 256, 256, 0, stream>>>(covh1, covW2, covb2, idx, span);
  // imp: base-only relu MLP -> sigmoid scalar
  gemm(comb, 1024, wImp1, DV, impb1, nullptr, imph1, 384, NP / 16, 384, DV, 1);
  dot384_sig_kernel<<<NP / 256, 256, 0, stream>>>(imph1, impW2, impb2, impv);
  // gpe tiny MLP
  gpe_kernel<<<NP / 256, 256, 0, stream>>>(idx, gpeW1, gpeb1, gpeW2, gpeb2, gtmp);
  // fuse
  fin_build_kernel<<<NP, 128, 0, stream>>>(comb, gtmp, rel_emb, wts, impv, span, fin);
  gemm(fin, 1184, wFuse1, 1184, fuseb1, fuse1f, nullptr, DV, NP / 16, DV, 1184, 0);
  ln_relu_kernel<<<NP, 256, 0, stream>>>(fuse1f, fuseg, fusebn, fuse1b, DV);
  gemm(fuse1b, DV, wFuse2, DV, fuseb2, out, nullptr, DV, NP / 16, DV, DV, 0);
}
